// CrossTransformer_score2_37520834298093
// MI455X (gfx1250) — matrix operand order repaired and checked by reading, not compiled, not run on hardware
//
#include <hip/hip_runtime.h>
#include <hip/hip_bf16.h>

typedef __bf16 bf16;
typedef __attribute__((ext_vector_type(16))) __bf16 v16bf;
typedef __attribute__((ext_vector_type(8)))  float  v8f;

#define THREADS 256
#define NWAVE   8

// problem dims
#define B_     64
#define C_     512
#define HW_    49
#define HWP_   64
#define DK_    128
#define MSUP_  1600
#define NCOL_  245    // n * hw = 5*49
#define NCOLP_ 256
#define SCALE_ 0.08838834764831845f   // 128^-0.5

// ---------------- WMMA helpers (wave32, 16x16x32 bf16 -> f32) ----------------

__device__ __forceinline__ v8f wmma_bf16(v16bf a, v16bf b, v8f c) {
  // (neg_a, A, neg_b, B, c_mod, C, reuse_a, reuse_b)
  return __builtin_amdgcn_wmma_f32_16x16x32_bf16(false, a, false, b, (short)0, c,
                                                 false, false);
}

// A fragment: row-major Amat[m][k], stride lda (elements). 16x32 bf16 tile.
// lanes 0-15: M=lane, K = 0..15 (contiguous)
// lanes 16-31: M=lane-16, K = 16..31
__device__ __forceinline__ v16bf load_frag_a(const bf16* A, int lda, int lane) {
  const int m  = lane & 15;
  const int kb = (lane >> 4) << 4;
  const bf16* p = A + m * lda + kb;
  v16bf a;
#pragma unroll
  for (int e = 0; e < 8; ++e) a[e] = p[e];
#pragma unroll
  for (int e = 0; e < 8; ++e) a[8 + e] = p[8 + e];
  return a;
}

// B fragment from TRANSPOSED storage Bt[n][k] (row-major n x k, stride ldb).
// 32x16 bf16 tile: lane n = lane&15, K = (lane>>4)*16 + 0..15 (contiguous).
__device__ __forceinline__ v16bf load_frag_bt(const bf16* Bt, int ldb, int lane) {
  const int n  = lane & 15;
  const int kb = (lane >> 4) << 4;
  const bf16* p = Bt + n * ldb + kb;
  v16bf b;
#pragma unroll
  for (int e = 0; e < 16; ++e) b[e] = p[e];
  return b;
}

// C/D f32 16x16: lane holds 8 rows: m = (lane>>4)*8 + i, n = lane&15

// ---------------- kernel 0: weight conversion fp32 -> bf16 ----------------
// layout in dst (elements): [0,65536) wqk | [65536,131072) wv
//                           [131072,196608) ws1 | [196608,200704) ws2
__global__ void k0_convert_weights(const float* __restrict__ wqk,
                                   const float* __restrict__ wv,
                                   const float* __restrict__ ws1,
                                   const float* __restrict__ ws2,
                                   bf16* __restrict__ dst) {
  int i = blockIdx.x * blockDim.x + threadIdx.x;
  if (i < 65536)        dst[i] = (bf16)wqk[i];
  else if (i < 131072)  dst[i] = (bf16)wv[i - 65536];
  else if (i < 196608)  dst[i] = (bf16)ws1[i - 131072];
  else if (i < 200704)  dst[i] = (bf16)ws2[i - 196608];
}

// ---------------- kernel 1: gating + sk/sv projection, one block per m ------
// smem: sBt [64][512] bf16 | t1t [64][64] bf16 | sfb [64][64] f32 |
//       avg [512] f32 | bias [64] f32
#define K1_SBT  0
#define K1_T1   65536
#define K1_SF   73728
#define K1_AVG  90112
#define K1_BIAS 92160
#define K1_SIZE 92416

__global__ __launch_bounds__(THREADS)
void k1_supports(const float* __restrict__ sup,
                 const bf16* __restrict__ wqk_bf, const bf16* __restrict__ wv_bf,
                 const bf16* __restrict__ ws1_bf, const bf16* __restrict__ ws2_bf,
                 bf16* __restrict__ sk_bf, bf16* __restrict__ sv_bf) {
  extern __shared__ char smem[];
  bf16*  sBt  = (bf16*)(smem + K1_SBT);    // [hw 64][c 512]  (B^T layout)
  bf16*  t1t  = (bf16*)(smem + K1_T1);     // [hw 64][o 64]
  float* sfb  = (float*)(smem + K1_SF);    // [o 64][hw 64]
  float* avg  = (float*)(smem + K1_AVG);   // [512]
  float* bias = (float*)(smem + K1_BIAS);  // [64]

  const int tid  = threadIdx.x;
  const int wave = tid >> 5;
  const int lane = tid & 31;
  const int m    = blockIdx.x;

  // phase 0: init + load s[m] transposed into LDS (bf16), accumulate avg
  for (int i = tid; i < 512 + 64; i += THREADS) avg[i] = 0.f;
  for (int i = tid; i < (HWP_ - HW_) * C_; i += THREADS)
    sBt[(HW_ + i / C_) * C_ + (i % C_)] = (bf16)0.f;
  __syncthreads();

  const float* sg = sup + (size_t)m * C_ * HW_;
  for (int f = tid; f < C_ * HW_; f += THREADS) {
    int c = f / HW_, hw = f % HW_;
    float v = sg[f];
    sBt[hw * C_ + c] = (bf16)v;
    atomicAdd(&avg[c], v);
  }
  __syncthreads();
  for (int c = tid; c < C_; c += THREADS) avg[c] *= (1.f / 49.f);
  __syncthreads();

  // phase 1: bias[o] = sum_c W1b[o][c] * avg[c]   (W1b = cols 512..1023)
  if (tid < 64) {
    float acc = 0.f;
    const bf16* wr = ws1_bf + tid * 1024 + 512;
    for (int c = 0; c < 512; ++c) acc += (float)wr[c] * avg[c];
    bias[tid] = acc;
  }
  __syncthreads();

  // phase 2: t1 = W1a (64x512) @ s (512x64) + bias  -> t1t (bf16, transposed)
  for (int t = wave; t < 16; t += NWAVE) {
    int mT = t >> 2, nT = t & 3;
    v8f acc = {};
    for (int kS = 0; kS < 16; ++kS) {
      v16bf a = load_frag_a(ws1_bf + (mT * 16) * 1024 + kS * 32, 1024, lane);
      v16bf b = load_frag_bt(sBt + (nT * 16) * C_ + kS * 32, C_, lane);
      acc = wmma_bf16(a, b, acc);
    }
    int n  = nT * 16 + (lane & 15);
    int mB = mT * 16 + ((lane >> 4) << 3);
#pragma unroll
    for (int i = 0; i < 8; ++i)
      t1t[n * 64 + mB + i] = (bf16)(acc[i] + bias[mB + i]);
  }
  __syncthreads();

  // phase 3: sf = W2 (64x64) @ t1 (64x64); sigmoid -> sfb
  for (int t = wave; t < 16; t += NWAVE) {
    int mT = t >> 2, nT = t & 3;
    v8f acc = {};
    for (int kS = 0; kS < 2; ++kS) {
      v16bf a = load_frag_a(ws2_bf + (mT * 16) * 64 + kS * 32, 64, lane);
      v16bf b = load_frag_bt(t1t + (nT * 16) * 64 + kS * 32, 64, lane);
      acc = wmma_bf16(a, b, acc);
    }
    int n  = nT * 16 + (lane & 15);
    int mB = mT * 16 + ((lane >> 4) << 3);
#pragma unroll
    for (int i = 0; i < 8; ++i)
      sfb[(mB + i) * HWP_ + n] = 1.f / (1.f + __expf(-acc[i]));
  }
  __syncthreads();

  // phase 4: softmax over the 64 channels, per spatial column
  if (tid < HW_) {
    float mx = -1e30f;
    for (int o = 0; o < 64; ++o) mx = fmaxf(mx, sfb[o * HWP_ + tid]);
    float s = 0.f;
    for (int o = 0; o < 64; ++o) {
      float e = __expf(sfb[o * HWP_ + tid] - mx);
      sfb[o * HWP_ + tid] = e;
      s += e;
    }
    float inv = 1.f / s;
    for (int o = 0; o < 64; ++o) sfb[o * HWP_ + tid] *= inv;
  }
  __syncthreads();

  // phase 5: ws = score[c%64] * s  (in place in sBt)
  for (int f = tid; f < HW_ * C_; f += THREADS) {
    int hw = f / C_, c = f % C_;
    float v = (float)sBt[hw * C_ + c] * sfb[(c & 63) * HWP_ + hw];
    sBt[hw * C_ + c] = (bf16)v;
  }
  __syncthreads();

  // phase 6: [sk;sv] = [Wqk;Wv] (256x512) @ ws (512x64)
  for (int t = wave; t < 64; t += NWAVE) {
    int mT = t >> 2, nT = t & 3;
    const bf16* Wp = (mT < 8) ? (wqk_bf + mT * 16 * 512)
                              : (wv_bf + (mT - 8) * 16 * 512);
    v8f acc = {};
    for (int kS = 0; kS < 16; ++kS) {
      v16bf a = load_frag_a(Wp + kS * 32, 512, lane);
      v16bf b = load_frag_bt(sBt + (nT * 16) * C_ + kS * 32, C_, lane);
      acc = wmma_bf16(a, b, acc);
    }
    int n = nT * 16 + (lane & 15);
    if (n < HW_) {
      int mB = (mT & 7) * 16 + ((lane >> 4) << 3);
      if (mT < 8) {  // sk: [m][hw][dk]
#pragma unroll
        for (int i = 0; i < 8; ++i)
          sk_bf[((size_t)m * HW_ + n) * DK_ + mB + i] = (bf16)acc[i];
      } else {       // sv: [m][dk][hw]
#pragma unroll
        for (int i = 0; i < 8; ++i)
          sv_bf[((size_t)m * DK_ + mB + i) * HW_ + n] = (bf16)acc[i];
      }
    }
  }
}

// ---------------- kernel 2: q / qv projection, one block per b --------------
__global__ __launch_bounds__(THREADS)
void k2_query(const float* __restrict__ query,
              const bf16* __restrict__ wqk_bf, const bf16* __restrict__ wv_bf,
              bf16* __restrict__ q_bf, float* __restrict__ qv_f) {
  extern __shared__ char smem[];
  bf16* qBt = (bf16*)smem;  // [hw 64][c 512]

  const int tid  = threadIdx.x;
  const int wave = tid >> 5;
  const int lane = tid & 31;
  const int b    = blockIdx.x;

  for (int i = tid; i < (HWP_ - HW_) * C_; i += THREADS)
    qBt[(HW_ + i / C_) * C_ + (i % C_)] = (bf16)0.f;
  __syncthreads();
  const float* qg = query + (size_t)b * C_ * HW_;
  for (int f = tid; f < C_ * HW_; f += THREADS) {
    int c = f / HW_, hw = f % HW_;
    qBt[hw * C_ + c] = (bf16)qg[f];
  }
  __syncthreads();

  for (int t = wave; t < 64; t += NWAVE) {
    int mT = t >> 2, nT = t & 3;
    const bf16* Wp = (mT < 8) ? (wqk_bf + mT * 16 * 512)
                              : (wv_bf + (mT - 8) * 16 * 512);
    v8f acc = {};
    for (int kS = 0; kS < 16; ++kS) {
      v16bf a = load_frag_a(Wp + kS * 32, 512, lane);
      v16bf b = load_frag_bt(qBt + (nT * 16) * C_ + kS * 32, C_, lane);
      acc = wmma_bf16(a, b, acc);
    }
    int n  = nT * 16 + (lane & 15);
    int mB = (mT & 7) * 16 + ((lane >> 4) << 3);
    if (mT < 8) {  // q: [b][hw(padded 64)][dk], pad rows zeroed
#pragma unroll
      for (int i = 0; i < 8; ++i)
        q_bf[(size_t)b * HWP_ * DK_ + n * DK_ + mB + i] =
            (n < HW_) ? (bf16)acc[i] : (bf16)0.f;
    } else if (n < HW_) {  // qv: [b][dk][hw], fp32 (exact for distance)
#pragma unroll
      for (int i = 0; i < 8; ++i)
        qv_f[((size_t)b * DK_ + mB + i) * HW_ + n] = acc[i];
    }
  }
}

// ---------------- kernel 3: attention + euclidean dist, one block per (b,k) -
// smem: Qs [64][128] bf16 | bufB 64KB (SKt[256][128] then SVt[128][256]) |
//       simS [64][256] f32 | attnA [64][256] bf16 | red [256] f32
#define K3_QS   0
#define K3_BUF  16384
#define K3_SIM  81920
#define K3_ATT  147456
#define K3_RED  180224
#define K3_SIZE 181248

__global__ __launch_bounds__(THREADS)
void k3_attention(const bf16* __restrict__ q_bf, const bf16* __restrict__ sk_bf,
                  const bf16* __restrict__ sv_bf, const float* __restrict__ qv_f,
                  float* __restrict__ out) {
  extern __shared__ char smem[];
  bf16*  Qs    = (bf16*)(smem + K3_QS);
  bf16*  bufB  = (bf16*)(smem + K3_BUF);
  float* simS  = (float*)(smem + K3_SIM);
  bf16*  attnA = (bf16*)(smem + K3_ATT);
  float* red   = (float*)(smem + K3_RED);

  const int tid  = threadIdx.x;
  const int wave = tid >> 5;
  const int lane = tid & 31;
  const int bk   = blockIdx.x;       // = b*5 + kk
  const int b    = bk / 5;
  const size_t m0 = (size_t)bk * 5;  // first support sample for this (b,k)

  // stage 1: Qs copy + SKt copy (contiguous: sk rows m0..m0+4 are [col][dk])
  for (int f = tid; f < HWP_ * DK_; f += THREADS)
    Qs[f] = q_bf[(size_t)b * HWP_ * DK_ + f];
  const bf16* skg = sk_bf + m0 * HW_ * DK_;
  for (int f = tid; f < NCOL_ * DK_; f += THREADS) bufB[f] = skg[f];
  for (int f = tid; f < (NCOLP_ - NCOL_) * DK_; f += THREADS)
    bufB[NCOL_ * DK_ + f] = (bf16)0.f;
  __syncthreads();

  // stage 2: sim = Q (64x128) @ SK^T (128x256), scaled
  for (int t = wave; t < 64; t += NWAVE) {
    int mT = t >> 4, nT = t & 15;
    v8f acc = {};
    for (int kS = 0; kS < 4; ++kS) {
      v16bf a = load_frag_a(Qs + (mT * 16) * DK_ + kS * 32, DK_, lane);
      v16bf bb = load_frag_bt(bufB + (nT * 16) * DK_ + kS * 32, DK_, lane);
      acc = wmma_bf16(a, bb, acc);
    }
    int n  = nT * 16 + (lane & 15);
    int mB = mT * 16 + ((lane >> 4) << 3);
#pragma unroll
    for (int i = 0; i < 8; ++i)
      simS[(size_t)(mB + i) * NCOLP_ + n] = acc[i] * SCALE_;
  }
  __syncthreads();

  // stage 3: overwrite bufB with SVt [dk 128][col 256]; zero attnA
  const bf16* svg = sv_bf + m0 * DK_ * HW_;
  for (int r = tid; r < 5 * DK_; r += THREADS) {
    int n = r / DK_, dk = r % DK_;
    const bf16* src = svg + ((size_t)n * DK_ + dk) * HW_;
    bf16* dst = bufB + dk * NCOLP_ + n * HW_;
    for (int j = 0; j < HW_; ++j) dst[j] = src[j];
  }
  for (int f = tid; f < DK_ * (NCOLP_ - NCOL_); f += THREADS) {
    int dk = f / (NCOLP_ - NCOL_);
    bufB[dk * NCOLP_ + NCOL_ + (f % (NCOLP_ - NCOL_))] = (bf16)0.f;
  }
  for (int f = tid; f < HWP_ * NCOLP_; f += THREADS) attnA[f] = (bf16)0.f;
  __syncthreads();

  // stage 4: row softmax over 245 keys -> attnA (bf16)
  if (tid < HW_) {
    const float* row = simS + (size_t)tid * NCOLP_;
    float mx = -1e30f;
    for (int j = 0; j < NCOL_; ++j) mx = fmaxf(mx, row[j]);
    float s = 0.f;
    for (int j = 0; j < NCOL_; ++j) s += __expf(row[j] - mx);
    float inv = 1.f / s;
    bf16* arow = attnA + (size_t)tid * NCOLP_;
    for (int j = 0; j < NCOL_; ++j) arow[j] = (bf16)(__expf(row[j] - mx) * inv);
  }
  __syncthreads();

  // stage 5: out = attn (64x256) @ SV (256x128), fused (qv-out)^2
  float locSum = 0.f;
  for (int t = wave; t < 32; t += NWAVE) {
    int mT = t >> 3, nT = t & 7;
    v8f acc = {};
    for (int kS = 0; kS < 8; ++kS) {
      v16bf a = load_frag_a(attnA + (mT * 16) * NCOLP_ + kS * 32, NCOLP_, lane);
      v16bf bb = load_frag_bt(bufB + (nT * 16) * NCOLP_ + kS * 32, NCOLP_, lane);
      acc = wmma_bf16(a, bb, acc);
    }
    int n  = nT * 16 + (lane & 15);              // dk
    int mB = mT * 16 + ((lane >> 4) << 3);       // hw
#pragma unroll
    for (int i = 0; i < 8; ++i) {
      int hw = mB + i;
      if (hw < HW_) {
        float d = qv_f[((size_t)b * DK_ + n) * HW_ + hw] - acc[i];
        locSum += d * d;
      }
    }
  }
  red[tid] = locSum;
  __syncthreads();
  for (int s = THREADS / 2; s > 0; s >>= 1) {
    if (tid < s) red[tid] += red[tid + s];
    __syncthreads();
  }
  if (tid == 0) out[bk] = -red[0] * (1.f / 49.f);
}

// ---------------- host launcher ----------------
extern "C" void kernel_launch(void* const* d_in, const int* in_sizes, int n_in,
                              void* d_out, int out_size, void* d_ws, size_t ws_size,
                              hipStream_t stream) {
  (void)in_sizes; (void)n_in; (void)out_size; (void)ws_size;
  const float* query = (const float*)d_in[0];
  const float* sup   = (const float*)d_in[1];
  const float* wqk   = (const float*)d_in[2];
  const float* wv    = (const float*)d_in[3];
  const float* ws1   = (const float*)d_in[4];
  const float* ws2   = (const float*)d_in[5];
  float* outp = (float*)d_out;

  // workspace layout (bytes)
  char* ws = (char*)d_ws;
  bf16* wqk_bf = (bf16*)(ws + 0);         // 128*512
  bf16* wv_bf  = (bf16*)(ws + 131072);    // 128*512
  bf16* ws1_bf = (bf16*)(ws + 262144);    // 64*1024
  bf16* ws2_bf = (bf16*)(ws + 393216);    // 64*64
  bf16* q_bf   = (bf16*)(ws + 401408);    // 64*64*128
  bf16* sk_bf  = (bf16*)(ws + 1449984);   // 1600*49*128  [m][hw][dk]
  bf16* sv_bf  = (bf16*)(ws + 21520384);  // 1600*128*49  [m][dk][hw]
  float* qv_f  = (float*)(ws + 41590784); // 64*128*49 fp32
  // total: 43,196,416 bytes

  k0_convert_weights<<<(200704 + 255) / 256, 256, 0, stream>>>(wqk, wv, ws1, ws2,
                                                               wqk_bf);
  k1_supports<<<MSUP_, THREADS, K1_SIZE, stream>>>(sup, wqk_bf, wv_bf, ws1_bf,
                                                   ws2_bf, sk_bf, sv_bf);
  k2_query<<<B_, THREADS, HWP_ * C_ * 2, stream>>>(query, wqk_bf, wv_bf, q_bf,
                                                   qv_f);
  k3_attention<<<B_ * 5, THREADS, K3_SIZE, stream>>>(q_bf, sk_bf, sv_bf, qv_f,
                                                     outp);
}
